// AdaptiveLinearWithChannel_81741817577981
// MI455X (gfx1250) — matrix operand order repaired and checked by reading, not compiled, not run on hardware
//
#include <hip/hip_runtime.h>
#include <hip/hip_bf16.h>
#include <stdint.h>

// ---------------------------------------------------------------------------
// CDNA5 (gfx1250) types for WMMA
// ---------------------------------------------------------------------------
typedef __attribute__((ext_vector_type(16))) __bf16 v16bf;
typedef __attribute__((ext_vector_type(8)))  float  v8f;

// bf16 split helpers: hi = truncate-to-bf16 (exact residual), lo = rne(x - hi)
__device__ __forceinline__ unsigned short bf16_trunc(float x) {
  union { float f; unsigned u; } v; v.f = x;
  return (unsigned short)(v.u >> 16);
}
__device__ __forceinline__ float bf16_up(unsigned short b) {
  union { float f; unsigned u; } v; v.u = ((unsigned)b) << 16;
  return v.f;
}
__device__ __forceinline__ unsigned short bf16_rne(float x) {
  union { float f; unsigned u; } v; v.f = x;
  unsigned u = v.u;
  unsigned r = u + 0x7FFFu + ((u >> 16) & 1u);
  return (unsigned short)(r >> 16);
}

// ---------------------------------------------------------------------------
// Kernel 1: hypernetwork + W_eff construction + B-fragment swizzle.
// One block per channel.  Builds:
//   addv[c][256]  : interleaved (bias + hyper_shift) add vector (f32)
//   bfrag         : Wcat (256x256) as bf16 hi/lo planes, pre-swizzled into
//                   v_wmma 16x16x32 bf16 B-matrix fragments (contiguous-K order, as A):
//                   [c][kk(8)][nt(16)][plane(2)][lane(32)][v(8)] dwords,
//                   lane = (n%16) + 16*(kc>=16), v packs K = 16h+2v, 16h+2v+1.
// Wcat folds the complex GEMM into one real GEMM with interleaved output:
//   Wcat[i][2o+0] = i<128 ?  Wr[i][o]      : -Wi[i-128][o]
//   Wcat[i][2o+1] = i<128 ?  Wi[i][o]      :  Wr[i-128][o]
// ---------------------------------------------------------------------------
__global__ void hyper_weff_setup(
    const float* __restrict__ weight, const float* __restrict__ bias,
    const float* __restrict__ t,
    const float* __restrict__ gW1, const float* __restrict__ gb1,
    const float* __restrict__ gW2, const float* __restrict__ gb2,
    const float* __restrict__ gW3, const float* __restrict__ gb3,
    const float* __restrict__ sW1, const float* __restrict__ sb1,
    const float* __restrict__ sW2, const float* __restrict__ sb2,
    const float* __restrict__ sW3, const float* __restrict__ sb3,
    const int* __restrict__ indices, const int* __restrict__ model_idx,
    float* __restrict__ addv, unsigned* __restrict__ bfrag)
{
  constexpr int C = 32, D = 128, H = 10;
  const int c   = blockIdx.x;
  const int ci  = indices[c];
  const int mi  = model_idx[0];
  const float tv = t[0];
  const int tid = threadIdx.x;

  __shared__ float uv[8 * D];   // 1024: [u_re | u_im | v_re | v_im] (D,2 each)
  __shared__ float sh[2 * D];   // 256 : hyper shift

  float h1[H], h2[H];

  // ---- gating MLP (1 -> 10 -> 10 -> 8D), redundantly per thread (tiny) ----
  for (int j = 0; j < H; ++j) { float a = tv * gW1[ci*H + j] + gb1[ci*H + j]; h1[j] = a > 0.f ? a : 0.f; }
  for (int j = 0; j < H; ++j) {
    float a = gb2[ci*H + j];
    for (int i = 0; i < H; ++i) a += h1[i] * gW2[(ci*H + i)*H + j];
    h2[j] = a > 0.f ? a : 0.f;
  }
  for (int o = tid; o < 8*D; o += blockDim.x) {
    float a = gb3[(size_t)ci*8*D + o];
    for (int i = 0; i < H; ++i) a += h2[i] * gW3[((size_t)ci*H + i)*8*D + o];
    uv[o] = a;
  }

  // ---- shift MLP (1 -> 10 -> 10 -> 2D) ----
  for (int j = 0; j < H; ++j) { float a = tv * sW1[ci*H + j] + sb1[ci*H + j]; h1[j] = a > 0.f ? a : 0.f; }
  for (int j = 0; j < H; ++j) {
    float a = sb2[ci*H + j];
    for (int i = 0; i < H; ++i) a += h1[i] * sW2[(ci*H + i)*H + j];
    h2[j] = a > 0.f ? a : 0.f;
  }
  for (int o = tid; o < 2*D; o += blockDim.x) {
    float a = sb3[(size_t)ci*2*D + o];
    for (int i = 0; i < H; ++i) a += h2[i] * sW3[((size_t)ci*H + i)*2*D + o];
    sh[o] = a;
  }
  __syncthreads();

  // ---- bias + hyper_shift add vector (interleaved re/im) ----
  if (tid < 2*D) {
    int o = tid >> 1, comp = tid & 1;
    float bv = bias[(((size_t)mi*C + c)*D + o)*2 + comp];
    float hv = comp ? sh[D + o] : sh[o];
    addv[c*2*D + tid] = bv + hv;
  }

  const float* wb = weight + (((size_t)mi*C + c)*D)*(size_t)D*2;

  // ---- build Wcat and emit bf16 hi/lo B fragments ----
  for (int idx = tid; idx < 128*256; idx += blockDim.x) {
    int kp = idx >> 8;        // K pair: rows i = 2kp, 2kp+1 of Wcat
    int j  = idx & 255;       // column of Wcat
    int o = j >> 1, comp = j & 1;
    float e[2];
    #pragma unroll
    for (int s = 0; s < 2; ++s) {
      int i  = 2*kp + s;
      int di = i & (D - 1);
      // rank-2 complex residual u[di] . v[o]^T
      float rr = 0.f, ri = 0.f;
      #pragma unroll
      for (int r = 0; r < 2; ++r) {
        float ur = uv[2*di + r],        ui = uv[2*D + 2*di + r];
        float vr = uv[4*D + 2*o + r],   vi = uv[6*D + 2*o + r];
        rr += ur*vr - ui*vi;
        ri += ur*vi + ui*vr;
      }
      float wr = wb[((size_t)di*D + o)*2 + 0] + rr;
      float wi = wb[((size_t)di*D + o)*2 + 1] + ri;
      e[s] = (i < D) ? (comp ? wi : wr) : (comp ? wr : -wi);
    }
    unsigned short h0  = bf16_trunc(e[0]);
    unsigned short h1b = bf16_trunc(e[1]);
    unsigned short l0  = bf16_rne(e[0] - bf16_up(h0));
    unsigned short l1b = bf16_rne(e[1] - bf16_up(h1b));

    int i0 = 2*kp;
    int kk = i0 >> 5, kc = i0 & 31;
    int h  = kc >> 4;                 // lane half selects K 0-15 / 16-31
    int v  = (kc & 15) >> 1;          // dword packs K = 16h+2v, 16h+2v+1
    int lane = (j & 15) + 16*h;
    int nt   = j >> 4;
    size_t base = ((((size_t)c*8 + kk)*16 + nt)*2 + 0)*256 + lane*8 + v;
    bfrag[base]       = ((unsigned)h1b << 16) | h0;   // hi plane
    bfrag[base + 256] = ((unsigned)l1b << 16) | l0;   // lo plane
  }
}

// ---------------------------------------------------------------------------
// Kernel 2: per-channel GEMM  Y(8192x256) = Xcat(8192x256) @ Wcat(256x256)
// via bf16 hi/lo split (3-term) v_wmma_f32_16x16x32_bf16.
// Block = 256 threads = 8 waves; tile 64 rows x 256 cols; K = 256 in 8 chunks.
// LDS: A fragments (64KB, staged once, non-temporal global reads) +
//      B fragment chunk (32KB per K-chunk, staged via
//      GLOBAL_LOAD_ASYNC_TO_LDS_B128 from L2-resident Wcat fragments,
//      shared by all 8 waves).  96KB < 320KB/WGP.
// ---------------------------------------------------------------------------
__global__ __launch_bounds__(256) void cgemm_weff(
    const float* __restrict__ xr, const float* __restrict__ xi,
    const float* __restrict__ addv, const unsigned* __restrict__ bfrag,
    float* __restrict__ out)
{
  constexpr int D = 128, P = 8192;
  extern __shared__ unsigned smem[];
  unsigned* lA = smem;            // 4 mt * 8 kk * 2 planes * 256 = 16384 dwords
  unsigned* lB = smem + 16384;    // 16 nt * 2 planes * 256      =  8192 dwords

  const int c   = blockIdx.y;
  const int rb  = blockIdx.x * 64;
  const int tid = threadIdx.x;

  // ---- stage X tile as hi/lo bf16 A fragments (contiguous-K layout, k = 16h+s) ----
  // X is streamed exactly once -> non-temporal loads (keep L2 for Wcat).
  for (int idx = tid; idx < 64*128; idx += 256) {
    int row = idx >> 7;           // 0..63
    int kp  = idx & 127;          // K pair
    int k   = kp << 1;
    const float* src = (k < D) ? (xr + (((size_t)c*P + rb + row)*D + k))
                               : (xi + (((size_t)c*P + rb + row)*D + (k - D)));
    float x0 = __builtin_nontemporal_load(src + 0);
    float x1 = __builtin_nontemporal_load(src + 1);
    unsigned short h0  = bf16_trunc(x0);
    unsigned short h1b = bf16_trunc(x1);
    unsigned short l0  = bf16_rne(x0 - bf16_up(h0));
    unsigned short l1b = bf16_rne(x1 - bf16_up(h1b));
    int mt = row >> 4, m = row & 15;
    int kk = k >> 5,  kc = k & 31;
    int h  = kc >> 4;                              // K in {16h..16h+15}
    int v  = (kc & 15) >> 1;                       // dword packs K = 16h+2v, 16h+2v+1
    int lane = m + 16*h;
    int base = ((mt*8 + kk)*2)*256 + lane*8 + v;
    lA[base]       = ((unsigned)h1b << 16) | h0;   // hi plane
    lA[base + 256] = ((unsigned)l1b << 16) | l0;   // lo plane
  }

  const int wid  = tid >> 5;      // wave32
  const int lane = tid & 31;
  const int wm   = wid & 3;       // row tile 0..3 (16 rows each)
  const int wn   = wid >> 2;      // column half 0..1 (8 col tiles each)

  v8f acc[8] = {};

  const unsigned* bsrc = bfrag + (size_t)c * 8 * 8192;

  // LDS byte offset of lB for the async-copy VDST operand
  // (generic LDS pointers carry the LDS offset in their low 32 bits).
  const unsigned lB_off = (unsigned)(uintptr_t)(void*)lB;

  for (int kk = 0; kk < 8; ++kk) {
    __syncthreads();              // all waves done with previous lB chunk
    // ---- async stage of this K-chunk's B fragments (32KB, contiguous) ----
    {
      const uint64_t srcbase = (uint64_t)(uintptr_t)(bsrc + (size_t)kk * 8192);
      #pragma unroll
      for (int q = 0; q < 8; ++q) {
        unsigned off = (unsigned)(tid * 16 + q * 4096);   // bytes
        unsigned dst = lB_off + off;
        asm volatile("global_load_async_to_lds_b128 %0, %1, %2"
                     :
                     : "v"(dst), "v"(off), "s"(srcbase)
                     : "memory");
      }
      asm volatile("s_wait_asynccnt 0x0" ::: "memory");
    }
    __syncthreads();              // async LDS writes visible to all waves

    v16bf ahi = *(const v16bf*)&lA[((wm*8 + kk)*2 + 0)*256 + lane*8];
    v16bf alo = *(const v16bf*)&lA[((wm*8 + kk)*2 + 1)*256 + lane*8];

    #pragma unroll
    for (int nt = 0; nt < 8; ++nt) {
      int ntg = wn*8 + nt;
      v16bf bhi = *(const v16bf*)&lB[(ntg*2 + 0)*256 + lane*8];
      v16bf blo = *(const v16bf*)&lB[(ntg*2 + 1)*256 + lane*8];
      // 3-term bf16 split: hi*hi + hi*lo + lo*hi  (~fp32 accuracy)
      acc[nt] = __builtin_amdgcn_wmma_f32_16x16x32_bf16(false, ahi, false, bhi,
                                                        (short)0, acc[nt], false, false);
      acc[nt] = __builtin_amdgcn_wmma_f32_16x16x32_bf16(false, ahi, false, blo,
                                                        (short)0, acc[nt], false, false);
      acc[nt] = __builtin_amdgcn_wmma_f32_16x16x32_bf16(false, alo, false, bhi,
                                                        (short)0, acc[nt], false, false);
    }
  }

  // ---- epilogue: add bias+shift, store interleaved-complex fp32 rows ----
  // Output is write-once streaming -> non-temporal stores.
  const int ncol  = lane & 15;
  const int halfm = lane >> 4;    // C/D layout: VGPR r -> M=r (+8 for lanes 16-31)
  #pragma unroll
  for (int nt = 0; nt < 8; ++nt) {
    int j = (wn*8 + nt)*16 + ncol;
    float av = addv[c*256 + j];
    #pragma unroll
    for (int r = 0; r < 8; ++r) {
      int row = rb + wm*16 + r + 8*halfm;
      __builtin_nontemporal_store(acc[nt][r] + av,
                                  out + ((size_t)c*P + row)*256 + j);
    }
  }
}

// ---------------------------------------------------------------------------
// Host launch
// ---------------------------------------------------------------------------
extern "C" void kernel_launch(void* const* d_in, const int* in_sizes, int n_in,
                              void* d_out, int out_size, void* d_ws, size_t ws_size,
                              hipStream_t stream) {
  (void)in_sizes; (void)n_in; (void)out_size; (void)ws_size;

  const float* x_real = (const float*)d_in[0];
  const float* x_imag = (const float*)d_in[1];
  const float* weight = (const float*)d_in[2];
  const float* bias   = (const float*)d_in[3];
  const float* t      = (const float*)d_in[4];
  const float* gW1 = (const float*)d_in[5];
  const float* gb1 = (const float*)d_in[6];
  const float* gW2 = (const float*)d_in[7];
  const float* gb2 = (const float*)d_in[8];
  const float* gW3 = (const float*)d_in[9];
  const float* gb3 = (const float*)d_in[10];
  const float* sW1 = (const float*)d_in[11];
  const float* sb1 = (const float*)d_in[12];
  const float* sW2 = (const float*)d_in[13];
  const float* sb2 = (const float*)d_in[14];
  const float* sW3 = (const float*)d_in[15];
  const float* sb3 = (const float*)d_in[16];
  const int* indices   = (const int*)d_in[17];
  const int* model_idx = (const int*)d_in[18];
  float* out = (float*)d_out;

  // workspace: [0,32KB) addv ; [32KB, 32KB+8MB) B fragments
  float*    addv  = (float*)d_ws;
  unsigned* bfrag = (unsigned*)((char*)d_ws + 32*1024);

  hipLaunchKernelGGL(hyper_weff_setup, dim3(32), dim3(256), 0, stream,
                     weight, bias, t,
                     gW1, gb1, gW2, gb2, gW3, gb3,
                     sW1, sb1, sW2, sb2, sW3, sb3,
                     indices, model_idx, addv, bfrag);

  hipLaunchKernelGGL(cgemm_weff, dim3(8192/64, 32), dim3(256), 96*1024, stream,
                     x_real, x_imag, addv, bfrag, out);
}
